// STGCN_54322746359875
// MI455X (gfx1250) — matrix operand order repaired and checked by reading, not compiled, not run on hardware
//
#include <hip/hip_runtime.h>
#include <hip/hip_bf16.h>

// ---------------------------------------------------------------------------
// STGCN forward for MI455X (gfx1250).
// Heavy GEMMs on v_wmma_f32_16x16x32_bf16 with bf16 hi/lo split (3 products)
// for ~fp32 accuracy at bf16 matrix-core rate.
//  - A_hat pre-split to planar bf16 hi/lo once; A fragments direct from global
//    (16 MB total -> L2-resident across all 22K blocks).
//  - Conv weights pre-transposed+padded to the exact B-fragment layout;
//    weight fragments direct from global (L0-hot).
//  - Register-blocked tiles: 6 WMMA/wave/k-step (gagg), 12 (timeblock).
// ---------------------------------------------------------------------------

typedef __attribute__((ext_vector_type(16))) __bf16 v16bf;
typedef __attribute__((ext_vector_type(8)))  float  v8f;

union Frag {
    v16bf v;
    uint4 q[2];
};

__device__ __forceinline__ unsigned short f2bf(float f) {
    unsigned u = __float_as_uint(f);
    u += 0x7FFFu + ((u >> 16) & 1u);          // round-to-nearest-even
    return (unsigned short)(u >> 16);
}
__device__ __forceinline__ float bf2f(unsigned short h) {
    return __uint_as_float(((unsigned)h) << 16);
}
__device__ __forceinline__ void split_bf(float f, unsigned short& hi, unsigned short& lo) {
    hi = f2bf(f);
    lo = f2bf(f - bf2f(hi));
}

__device__ __forceinline__ v8f wmma_bf16(const Frag& a, const Frag& b, v8f c) {
    return __builtin_amdgcn_wmma_f32_16x16x32_bf16(
        false, a.v, false, b.v, (short)0, c, false, false);
}

__device__ __forceinline__ Frag ld_frag_g(const unsigned short* p) {
    Frag f;
    f.q[0] = *(const uint4*)p;
    f.q[1] = *(const uint4*)(p + 8);
    return f;
}

// ---------------------------------------------------------------------------
// float -> planar bf16 hi/lo split
// ---------------------------------------------------------------------------
__global__ __launch_bounds__(256) void split_kernel(
    const float* __restrict__ src,
    unsigned short* __restrict__ h, unsigned short* __restrict__ l, int n)
{
    int i = blockIdx.x * 256 + threadIdx.x;
    if (i >= n) return;
    unsigned short hh, ll;
    split_bf(src[i], hh, ll);
    h[i] = hh;
    l[i] = ll;
}

// ---------------------------------------------------------------------------
// Build transposed, K-padded, bf16-split combined conv weights:
//   WcT[j][kp], j in [0,128): j<64 -> W0+W1 (temp), j>=64 -> W2 (gate)
//   kp in [0,KP) padded with zeros beyond KW=3*Cin.  Row pitch = KP.
// ---------------------------------------------------------------------------
__global__ __launch_bounds__(256) void prep_wcT(
    const float* __restrict__ W,           // [3,1,3,Cin,64]
    unsigned short* __restrict__ Th, unsigned short* __restrict__ Tl,
    int Cin, int KP)
{
    int idx = blockIdx.x * 256 + threadIdx.x;
    if (idx >= 128 * KP) return;
    int j  = idx / KP;
    int kp = idx - j * KP;
    int KW = 3 * Cin;
    float v = 0.f;
    if (kp < KW) {
        int k   = kp / Cin;
        int cin = kp - k * Cin;
        size_t base = ((size_t)k * Cin + cin) * 64;
        size_t cstr = (size_t)3 * Cin * 64;
        v = (j < 64) ? (W[base + j] + W[cstr + base + j])
                     : W[2 * cstr + base + (j - 64)];
    }
    unsigned short hh, ll;
    split_bf(v, hh, ll);
    Th[idx] = hh;
    Tl[idx] = ll;
}

// ---------------------------------------------------------------------------
// Graph aggregation:  Y[b,i,c] = sum_j A[i,j] * X[b,j,c]
// Block tile 64(M) x 64(N); wave = 16x32 (2 accumulators).
// A fragments: direct global loads from pre-split bf16 planes (L2-hot).
// X tile: fp32 -> split -> transposed LDS.
// ---------------------------------------------------------------------------
__global__ __launch_bounds__(256) void gagg_wmma(
    const unsigned short* __restrict__ Ah,   // [Nn,Nn] bf16 hi
    const unsigned short* __restrict__ Al,   // [Nn,Nn] bf16 lo
    const float* __restrict__ X,             // [B, Nn, TC]
    float* __restrict__ Y,                   // [B, Nn, TC]
    int Nn, int TC)
{
    __shared__ unsigned short sBh[64][40];
    __shared__ unsigned short sBl[64][40];

    const int tid  = threadIdx.x;
    const int lane = tid & 31;
    const int wave = tid >> 5;
    const int wm   = wave & 3;            // M sub-tile
    const int wn   = wave >> 2;           // N half (32 cols each)

    const int m0 = blockIdx.x * 64;
    const int n0 = blockIdx.y * 64;
    const size_t bofs = (size_t)blockIdx.z * (size_t)Nn * (size_t)TC;
    const float* Xb = X + bofs;
    float*       Yb = Y + bofs;

    const int xr = tid >> 3;              // 0..31 (K row of X tile)
    const int xc = (tid & 7) * 8;         // 0..56

    v8f acc0 = {0.f, 0.f, 0.f, 0.f, 0.f, 0.f, 0.f, 0.f};
    v8f acc1 = {0.f, 0.f, 0.f, 0.f, 0.f, 0.f, 0.f, 0.f};

    // per-lane A row pointers (contiguous k order: lane half h holds k = 16h + s)
    const int arow = m0 + wm * 16 + (lane & 15);
    const int akb  = (lane < 16) ? 0 : 16;
    const unsigned short* aph = Ah + (size_t)arow * Nn + akb;
    const unsigned short* apl = Al + (size_t)arow * Nn + akb;

    const int brow = wn * 32 + (lane & 15);
    const int bkb  = (lane < 16) ? 0 : 16;

    for (int k0 = 0; k0 < Nn; k0 += 32) {
        // ---- stage X tile (32K x 64N fp32 -> bf16 hi/lo, transposed) ----
        const float* xp = Xb + (size_t)(k0 + xr) * TC + (n0 + xc);
        float xv[8];
        *(float4*)&xv[0] = *(const float4*)xp;
        *(float4*)&xv[4] = *(const float4*)(xp + 4);
        if (k0 + 32 < Nn) __builtin_prefetch(xp + 32 * TC, 0, 3);
#pragma unroll
        for (int e = 0; e < 8; ++e) {
            unsigned short h, l; split_bf(xv[e], h, l);
            sBh[xc + e][xr] = h;
            sBl[xc + e][xr] = l;
        }
        __syncthreads();

        Frag fah = ld_frag_g(aph + k0);
        Frag fal = ld_frag_g(apl + k0);

        Frag fb0h, fb0l, fb1h, fb1l;
        fb0h.q[0] = *(const uint4*)&sBh[brow][bkb];
        fb0h.q[1] = *(const uint4*)&sBh[brow][bkb + 8];
        fb0l.q[0] = *(const uint4*)&sBl[brow][bkb];
        fb0l.q[1] = *(const uint4*)&sBl[brow][bkb + 8];
        fb1h.q[0] = *(const uint4*)&sBh[brow + 16][bkb];
        fb1h.q[1] = *(const uint4*)&sBh[brow + 16][bkb + 8];
        fb1l.q[0] = *(const uint4*)&sBl[brow + 16][bkb];
        fb1l.q[1] = *(const uint4*)&sBl[brow + 16][bkb + 8];

        acc0 = wmma_bf16(fah, fb0h, acc0);
        acc0 = wmma_bf16(fah, fb0l, acc0);
        acc0 = wmma_bf16(fal, fb0h, acc0);
        acc1 = wmma_bf16(fah, fb1h, acc1);
        acc1 = wmma_bf16(fah, fb1l, acc1);
        acc1 = wmma_bf16(fal, fb1h, acc1);
        __syncthreads();
    }

    const int col0 = n0 + wn * 32 + (lane & 15);
    const int r0   = m0 + wm * 16 + ((lane < 16) ? 0 : 8);
#pragma unroll
    for (int v = 0; v < 8; ++v) {
        Yb[(size_t)(r0 + v) * TC + col0]      = acc0[v];
        Yb[(size_t)(r0 + v) * TC + col0 + 16] = acc1[v];
    }
}

// ---------------------------------------------------------------------------
// Gated temporal conv (implicit-im2col GEMM) with fused epilogue.
// Block: 64 rows x all 128 cols (64 temp + 64 gate). Wave: 16 rows x 32 cols,
// 4 accumulators (2 temp tiles + 2 gate tiles) -> 12 WMMA per k-step.
// Weight fragments direct from global (pre-transposed split layout).
// ---------------------------------------------------------------------------
__global__ __launch_bounds__(256) void timeblock_wmma(
    const float* __restrict__ X,             // [NB, T_in, Cin]
    const unsigned short* __restrict__ WTh,  // [128, KP] bf16 hi (transposed)
    const unsigned short* __restrict__ WTl,  // [128, KP] bf16 lo
    const float* __restrict__ bias,          // [3, 64]
    float* __restrict__ Out,                 // [NB, T_out, 64]
    int T_in, int Cin, int T_out, int KP)
{
    const int KW = 3 * Cin;
    __shared__ unsigned short sAh[64][40];
    __shared__ unsigned short sAl[64][40];

    const int tid  = threadIdx.x;
    const int lane = tid & 31;
    const int wave = tid >> 5;
    const int wm   = wave & 3;
    const int wn   = wave >> 2;

    const int row0 = blockIdx.x * 64;
    const int ar   = tid >> 2;
    const int ac   = (tid & 3) * 8;

    v8f aT0 = {0.f,0.f,0.f,0.f,0.f,0.f,0.f,0.f};
    v8f aT1 = {0.f,0.f,0.f,0.f,0.f,0.f,0.f,0.f};
    v8f aG0 = {0.f,0.f,0.f,0.f,0.f,0.f,0.f,0.f};
    v8f aG1 = {0.f,0.f,0.f,0.f,0.f,0.f,0.f,0.f};

    // per-lane weight row pointers (B-fragment layout: 16 contiguous bf16)
    const int c0  = wn * 32 + (lane & 15);
    const int bkb = (lane < 16) ? 0 : 16;
    const unsigned short* wt0h = WTh + (size_t)c0 * KP + bkb;
    const unsigned short* wt1h = WTh + (size_t)(c0 + 16) * KP + bkb;
    const unsigned short* wg0h = WTh + (size_t)(c0 + 64) * KP + bkb;
    const unsigned short* wg1h = WTh + (size_t)(c0 + 80) * KP + bkb;
    const unsigned short* wt0l = WTl + (size_t)c0 * KP + bkb;
    const unsigned short* wt1l = WTl + (size_t)(c0 + 16) * KP + bkb;
    const unsigned short* wg0l = WTl + (size_t)(c0 + 64) * KP + bkb;
    const unsigned short* wg1l = WTl + (size_t)(c0 + 80) * KP + bkb;

    // per-thread staging row
    const int r  = row0 + ar;
    const int nd = r / T_out;
    const int t0 = r - nd * T_out;
    const float* rp = X + ((size_t)nd * T_in + t0) * Cin;

    const int arow = wm * 16 + (lane & 15);
    const int akb  = (lane < 16) ? 0 : 16;

    for (int k0 = 0; k0 < KP; k0 += 32) {
        // ---- stage data rows (implicit im2col: 3*Cin contiguous floats) ----
#pragma unroll
        for (int e = 0; e < 8; ++e) {
            int   kp = k0 + ac + e;
            float v  = (kp < KW) ? rp[kp] : 0.f;
            unsigned short h, l; split_bf(v, h, l);
            sAh[ar][ac + e] = h;
            sAl[ar][ac + e] = l;
        }
        __syncthreads();

        Frag fah, fal;
        fah.q[0] = *(const uint4*)&sAh[arow][akb];
        fah.q[1] = *(const uint4*)&sAh[arow][akb + 8];
        fal.q[0] = *(const uint4*)&sAl[arow][akb];
        fal.q[1] = *(const uint4*)&sAl[arow][akb + 8];

        Frag ft0h = ld_frag_g(wt0h + k0), ft0l = ld_frag_g(wt0l + k0);
        Frag ft1h = ld_frag_g(wt1h + k0), ft1l = ld_frag_g(wt1l + k0);
        Frag fg0h = ld_frag_g(wg0h + k0), fg0l = ld_frag_g(wg0l + k0);
        Frag fg1h = ld_frag_g(wg1h + k0), fg1l = ld_frag_g(wg1l + k0);

        aT0 = wmma_bf16(fah, ft0h, aT0);
        aT0 = wmma_bf16(fah, ft0l, aT0);
        aT0 = wmma_bf16(fal, ft0h, aT0);
        aT1 = wmma_bf16(fah, ft1h, aT1);
        aT1 = wmma_bf16(fah, ft1l, aT1);
        aT1 = wmma_bf16(fal, ft1h, aT1);
        aG0 = wmma_bf16(fah, fg0h, aG0);
        aG0 = wmma_bf16(fah, fg0l, aG0);
        aG0 = wmma_bf16(fal, fg0h, aG0);
        aG1 = wmma_bf16(fah, fg1h, aG1);
        aG1 = wmma_bf16(fah, fg1l, aG1);
        aG1 = wmma_bf16(fal, fg1h, aG1);
        __syncthreads();
    }

    // ---- fused gate epilogue: relu(sigmoid(g+bg) * (t+bt)) ----
    const int   m0c = wn * 32 + (lane & 15);
    const int   m1c = m0c + 16;
    const float bt0 = bias[m0c] + bias[64 + m0c];
    const float bg0 = bias[128 + m0c];
    const float bt1 = bias[m1c] + bias[64 + m1c];
    const float bg1 = bias[128 + m1c];
    const int   r0  = row0 + wm * 16 + ((lane < 16) ? 0 : 8);
#pragma unroll
    for (int v = 0; v < 8; ++v) {
        float t = aT0[v] + bt0;
        float g = aG0[v] + bg0;
        float s = 1.f / (1.f + __expf(-g));
        Out[(size_t)(r0 + v) * 64 + m0c] = fmaxf(s * t, 0.f);
        t = aT1[v] + bt1;
        g = aG1[v] + bg1;
        s = 1.f / (1.f + __expf(-g));
        Out[(size_t)(r0 + v) * 64 + m1c] = fmaxf(s * t, 0.f);
    }
}

// ---------------------------------------------------------------------------
// t2 = relu(X[rows,64] @ theta[64,16])
// ---------------------------------------------------------------------------
__global__ __launch_bounds__(256) void theta_relu(
    const float* __restrict__ X, const float* __restrict__ Th,
    float* __restrict__ Out, int Rows)
{
    __shared__ float sT[1024];
    for (int i = threadIdx.x; i < 1024; i += 256) sT[i] = Th[i];
    __syncthreads();
    int r = blockIdx.x * 256 + threadIdx.x;
    if (r >= Rows) return;
    const float* xp = X + (size_t)r * 64;
    float s[16];
#pragma unroll
    for (int p = 0; p < 16; ++p) s[p] = 0.f;
    for (int q = 0; q < 64; ++q) {
        float xq = xp[q];
#pragma unroll
        for (int p = 0; p < 16; ++p) s[p] += xq * sT[q * 16 + p];
    }
#pragma unroll
    for (int p = 0; p < 16; ++p)
        Out[(size_t)r * 16 + p] = fmaxf(s[p], 0.f);
}

// ---------------------------------------------------------------------------
// Final FC: out[row, p] = X[row, 0:896] @ Wf[896,12] + bf
// ---------------------------------------------------------------------------
__global__ __launch_bounds__(256) void fc_kernel(
    const float* __restrict__ X, const float* __restrict__ Wf,
    const float* __restrict__ bf, float* __restrict__ Out, int Rows)
{
    __shared__ float sW[10752];
    for (int i = threadIdx.x; i < 10752; i += 256) sW[i] = Wf[i];
    __syncthreads();
    int r = blockIdx.x * 256 + threadIdx.x;
    if (r >= Rows) return;
    const float* xp = X + (size_t)r * 896;
    float s[12];
#pragma unroll
    for (int p = 0; p < 12; ++p) s[p] = bf[p];
    for (int q = 0; q < 896; ++q) {
        float xq = xp[q];
#pragma unroll
        for (int p = 0; p < 12; ++p) s[p] += xq * sW[q * 12 + p];
    }
#pragma unroll
    for (int p = 0; p < 12; ++p) Out[(size_t)r * 12 + p] = s[p];
}

// ---------------------------------------------------------------------------
extern "C" void kernel_launch(void* const* d_in, const int* in_sizes, int n_in,
                              void* d_out, int out_size, void* d_ws, size_t ws_size,
                              hipStream_t stream)
{
    const float* x       = (const float*)d_in[0];
    const float* A_hat   = (const float*)d_in[1];
    const float* tb1_w   = (const float*)d_in[2];
    const float* tb1_b   = (const float*)d_in[3];
    const float* theta1  = (const float*)d_in[4];
    const float* tb2_w   = (const float*)d_in[5];
    const float* tb2_b   = (const float*)d_in[6];
    const float* tb3_w   = (const float*)d_in[7];
    const float* tb3_b   = (const float*)d_in[8];
    const float* theta2  = (const float*)d_in[9];
    const float* tb4_w   = (const float*)d_in[10];
    const float* tb4_b   = (const float*)d_in[11];
    const float* tb5_w   = (const float*)d_in[12];
    const float* tb5_b   = (const float*)d_in[13];
    const float* fully_w = (const float*)d_in[14];
    const float* fully_b = (const float*)d_in[15];

    const int Bb = 16, Nn = 2048;
    const int NB = Bb * Nn;                       // 32768 nodes
    const size_t sizeA = (size_t)NB * 22 * 64;    // max fp32 intermediate

    float* bufA = (float*)d_ws;
    float* bufB = bufA + sizeA;
    unsigned short* Ah  = (unsigned short*)(bufB + sizeA);
    unsigned short* Al  = Ah + (size_t)Nn * Nn;
    unsigned short* wTh = Al + (size_t)Nn * Nn;
    unsigned short* wTl = wTh + (size_t)128 * 192;

    // --- one-time A_hat bf16 hi/lo split (L2-resident: 16 MB) ---
    split_kernel<<<(Nn * Nn + 255) / 256, 256, 0, stream>>>(A_hat, Ah, Al, Nn * Nn);

    // --- block 1 ---
    prep_wcT<<<(128 * 32 + 255) / 256, 256, 0, stream>>>(tb1_w, wTh, wTl, 2, 32);
    timeblock_wmma<<<dim3(NB * 22 / 64, 1, 1), 256, 0, stream>>>(
        x, wTh, wTl, tb1_b, bufA, 24, 2, 22, 32);
    gagg_wmma<<<dim3(Nn / 64, (22 * 64) / 64, Bb), 256, 0, stream>>>(
        Ah, Al, bufA, bufB, Nn, 22 * 64);
    theta_relu<<<(NB * 22 + 255) / 256, 256, 0, stream>>>(
        bufB, theta1, bufA, NB * 22);
    prep_wcT<<<(128 * 64 + 255) / 256, 256, 0, stream>>>(tb2_w, wTh, wTl, 16, 64);
    timeblock_wmma<<<dim3(NB * 20 / 64, 1, 1), 256, 0, stream>>>(
        bufA, wTh, wTl, tb2_b, bufB, 22, 16, 20, 64);

    // --- block 2 ---
    prep_wcT<<<(128 * 192 + 255) / 256, 256, 0, stream>>>(tb3_w, wTh, wTl, 64, 192);
    timeblock_wmma<<<dim3(NB * 18 / 64, 1, 1), 256, 0, stream>>>(
        bufB, wTh, wTl, tb3_b, bufA, 20, 64, 18, 192);
    gagg_wmma<<<dim3(Nn / 64, (18 * 64) / 64, Bb), 256, 0, stream>>>(
        Ah, Al, bufA, bufB, Nn, 18 * 64);
    theta_relu<<<(NB * 18 + 255) / 256, 256, 0, stream>>>(
        bufB, theta2, bufA, NB * 18);
    prep_wcT<<<(128 * 64 + 255) / 256, 256, 0, stream>>>(tb4_w, wTh, wTl, 16, 64);
    timeblock_wmma<<<dim3(NB * 16 / 64, 1, 1), 256, 0, stream>>>(
        bufA, wTh, wTl, tb4_b, bufB, 18, 16, 16, 64);

    // --- final timeblock + FC ---
    prep_wcT<<<(128 * 192 + 255) / 256, 256, 0, stream>>>(tb5_w, wTh, wTl, 64, 192);
    timeblock_wmma<<<dim3(NB * 14 / 64, 1, 1), 256, 0, stream>>>(
        bufB, wTh, wTl, tb5_b, bufA, 16, 64, 14, 192);

    float* out4 = (float*)d_out;
    fc_kernel<<<(NB + 255) / 256, 256, 0, stream>>>(
        bufA, fully_w, fully_b, out4, NB);

    // second tuple element: A_hat passthrough
    hipMemcpyAsync(out4 + (size_t)NB * 12, A_hat,
                   (size_t)Nn * Nn * sizeof(float),
                   hipMemcpyDeviceToDevice, stream);
}
